// Encoder_10608569221395
// MI455X (gfx1250) — matrix operand order repaired and checked by reading, not compiled, not run on hardware
//
#include <hip/hip_runtime.h>
#include <math.h>

#define TT    40
#define NK    641
#define KP    656      // k rows padded to 41*16
#define JP    672      // inner dim padded to 21*32
#define MM    150
#define MR    160      // m padded to 10*16
#define NB    256
#define MAXIT 100
#define MP    168      // LDS Y column stride (mult of 8 -> 16B rows)
#define APC   680      // LDS A-slab column stride (mult of 8)
#define NTH   640      // 20 waves
#define NST   21       // ceil(41 ktiles / 2 per stage)
#define SLABW (32 * APC)

typedef __bf16 v16bf __attribute__((ext_vector_type(16)));
typedef float  v8f   __attribute__((ext_vector_type(8)));
typedef unsigned int uint4v __attribute__((ext_vector_type(4)));

__device__ __forceinline__ float bf2f(unsigned short u) {
    unsigned int x = ((unsigned int)u) << 16;
    float f; __builtin_memcpy(&f, &x, 4); return f;
}
__device__ __forceinline__ unsigned short f2bf(float f) {
    unsigned int x; __builtin_memcpy(&x, &f, 4);
    x += 0x7fffu + ((x >> 16) & 1u);                 // RNE
    return (unsigned short)(x >> 16);
}

union Frag16 { v16bf v; uint4v q[2]; };
union Acc8   { v8f v; float f[8]; };
union Q8     { uint4v q; unsigned short h[8]; };

// ---- CDNA5 async global->LDS DMA (ASYNCcnt path, ISA §15.18 op 98) ----
__device__ __forceinline__ void async_ld_b128(const unsigned short* gsrc, unsigned short* ldst) {
    unsigned lds_off = (unsigned)(size_t)ldst;            // generic LDS addr low 32b = wave-relative offset
    unsigned long long ga = (unsigned long long)(size_t)gsrc;
    asm volatile("global_load_async_to_lds_b128 %0, %1, off"
                 :: "v"(lds_off), "v"(ga) : "memory");
}
__device__ __forceinline__ void wait_async0() {
#if __has_builtin(__builtin_amdgcn_s_wait_asynccnt)
    __builtin_amdgcn_s_wait_asynccnt(0);
#else
    asm volatile("s_wait_asynccnt 0x0" ::: "memory");
#endif
}

// ---------------- setup kernels ----------------

__global__ void k_build_D(const float* __restrict__ Drr, const float* __restrict__ Dth,
                          float* __restrict__ D) {
    int k = threadIdx.x;
    if (k >= KP) return;
    if (k >= NK) { for (int i = 0; i < TT; ++i) D[i * KP + k] = 0.f; return; }
    float vals[TT];
    if (k == 0) {
        for (int i = 0; i < TT; ++i) vals[i] = 1.f;
    } else {
        int g = (k - 1) / 160, n = (k - 1) % 160;
        float r = Drr[n], th = Dth[n];
        float p = 1.f;
        for (int i = 0; i < TT; ++i) {
            float s, c; sincosf((float)i * th, &s, &c);
            float sg = (i & 1) ? -1.f : 1.f;
            float v = (g == 0) ? p * c : (g == 1) ? sg * p * c : (g == 2) ? p * s : sg * p * s;
            vals[i] = v; p *= r;
        }
    }
    float n2 = 0.f;
    for (int i = 0; i < TT; ++i) n2 += vals[i] * vals[i];
    float nrm = sqrtf(n2);
    if (nrm == 0.f) nrm = sqrtf((float)TT);
    for (int i = 0; i < TT; ++i) D[i * KP + k] = vals[i] / nrm;
}

__global__ void k_DtD(const float* __restrict__ D, float* __restrict__ DtD) {
    int idx = blockIdx.x * blockDim.x + threadIdx.x;
    if (idx >= NK * NK) return;
    int k = idx / NK, j = idx % NK;
    float s = 0.f;
    for (int t = 0; t < TT; ++t) s += D[t * KP + k] * D[t * KP + j];
    DtD[idx] = s;
}

__global__ void k_power(const float* __restrict__ DtD, float* __restrict__ scal) {
    __shared__ float v[NK];
    __shared__ float red[1024];
    int t = threadIdx.x;
    if (t < NK) v[t] = 1.f;
    __syncthreads();
    float nrm = 1.f;
    for (int it = 0; it < 64; ++it) {
        float w = 0.f;
        if (t < NK) {
            const float* row = DtD + (size_t)t * NK;
            for (int j = 0; j < NK; ++j) w += row[j] * v[j];
        }
        red[t] = (t < NK) ? w * w : 0.f;
        __syncthreads();
        for (int s = 512; s > 0; s >>= 1) { if (t < s) red[t] += red[t + s]; __syncthreads(); }
        nrm = sqrtf(red[0]);
        __syncthreads();
        if (t < NK) v[t] = w / nrm;
        __syncthreads();
    }
    if (t == 0) { scal[0] = nrm; scal[1] = 1.f / nrm; scal[2] = 0.1f / nrm; }
}

__global__ void k_buildA(const float* __restrict__ DtD, const float* __restrict__ scal,
                         unsigned short* __restrict__ Abf) {
    int idx = blockIdx.x * blockDim.x + threadIdx.x;
    if (idx >= KP * JP) return;
    int k = idx / JP, j = idx % JP;
    float val = 0.f;
    if (k < NK && j < NK) val = ((k == j) ? 1.f : 0.f) - DtD[k * NK + j] * scal[1];
    Abf[idx] = f2bf(val);
}

__global__ void k_DtY(const float* __restrict__ X, const float* __restrict__ D,
                      const float* __restrict__ scal, unsigned short* __restrict__ DtY) {
    int b = blockIdx.x / 10, mt = blockIdx.x % 10;
    __shared__ float xs[TT * 16];
    int t = threadIdx.x;
    for (int e = t; e < TT * 16; e += blockDim.x) {
        int ti = e >> 4, c = e & 15, m = mt * 16 + c;
        xs[e] = (m < MM) ? X[((size_t)b * TT + ti) * MM + m] : 0.f;
    }
    __syncthreads();
    float linv = scal[1];
    for (int p = t; p < KP * 16; p += blockDim.x) {
        int k = p >> 4, c = p & 15;
        float s = 0.f;
        for (int ti = 0; ti < TT; ++ti) s += D[ti * KP + k] * xs[ti * 16 + c];
        int m = mt * 16 + c;
        DtY[((size_t)b * MR + m) * KP + k] = f2bf(s * linv);
    }
}

// ---------------- main FISTA kernel: 1 persistent workgroup per batch ----------------

__global__ void __launch_bounds__(NTH, 1)
k_fista(const unsigned short* __restrict__ Abf, const unsigned short* __restrict__ DtY,
        unsigned short* __restrict__ Xa, unsigned short* __restrict__ Xb,
        const float* __restrict__ scal) {
    __shared__ unsigned short Ylds[JP * MP];       // y operand, resident whole kernel (~221 KB)
    __shared__ unsigned short Aslab[2 * SLABW];    // double-buffered A slab (2 k-tiles) (~87 KB)

    const int tid  = threadIdx.x;
    const int b    = blockIdx.x;
    const int wave = tid >> 5, lane = tid & 31;
    const int lhi  = lane >> 4, lm = lane & 15;
    const int ktl  = (wave >= 10) ? 1 : 0;         // which k-tile of the stage this wave owns
    const int mt   = wave - ktl * 10;              // which n-tile (m) this wave owns
    const float lam = scal[2];

    // y0 = 0 (also zeroes j-pad rows); x_prev = 0
    for (int i = tid; i < JP * MP; i += NTH) Ylds[i] = 0;
    {
        unsigned int* xb32 = (unsigned int*)(Xb + (size_t)b * MR * KP);
        for (int i = tid; i < MR * KP / 2; i += NTH) xb32[i] = 0u;
    }
    __syncthreads();

    float tcur = 1.f;
    for (int it = 0; it < MAXIT; ++it) {
        float tnew = 0.5f * (1.f + sqrtf(1.f + 4.f * tcur * tcur));
        float tt   = (tcur - 1.f) / tnew;
        tcur = tnew;
        unsigned short*       XnB = ((it & 1) ? Xb : Xa) + (size_t)b * MR * KP;
        const unsigned short* XoB = ((it & 1) ? Xa : Xb) + (size_t)b * MR * KP;

        // -------- phase A: Z = A @ Y, x_new = shrink(Z + DtY) --------
        // async-DMA stage 0 into slab buffer 0
        #pragma unroll
        for (int l = 0; l < 5; ++l) {
            int c = tid + l * NTH;
            if (c < 2688) {
                int row = c / 84, col = (c % 84) * 8;
                async_ld_b128(Abf + (size_t)row * JP + col, Aslab + row * APC + col);
            }
        }
        wait_async0();
        __syncthreads();

        for (int s = 0; s < NST; ++s) {
            // issue async DMA for stage s+1 into the other slab buffer
            if (s + 1 < NST) {
                unsigned short* nbuf = Aslab + ((s + 1) & 1) * SLABW;
                #pragma unroll
                for (int l = 0; l < 5; ++l) {
                    int c = tid + l * NTH;
                    if (c < 2688) {
                        int row = c / 84, col = (c % 84) * 8;
                        int grow = (s + 1) * 32 + row;
                        if (grow < KP)
                            async_ld_b128(Abf + (size_t)grow * JP + col, nbuf + row * APC + col);
                    }
                }
            }
            // compute stage s from current slab buffer
            int kt = s * 2 + ktl;
            if (kt < 41) {
                Acc8 acc;
                #pragma unroll
                for (int r = 0; r < 8; ++r) acc.f[r] = 0.f;
                const unsigned short* arow  = Aslab + (s & 1) * SLABW + (ktl * 16 + lm) * APC + lhi * 8;
                const unsigned short* bbase = Ylds + (size_t)(lhi * 8) * MP + mt * 16 + lm;
                for (int js = 0; js < 21; ++js) {
                    Frag16 fa, fb;
                    fa.q[0] = *(const uint4v*)(arow + js * 32);        // K base..base+7
                    fa.q[1] = *(const uint4v*)(arow + js * 32 + 16);   // K base+16..+23
                    Q8 b0, b1;
                    #pragma unroll
                    for (int r = 0; r < 8; ++r) {
                        b0.h[r] = bbase[(size_t)(js * 32 + r) * MP];
                        b1.h[r] = bbase[(size_t)(js * 32 + 16 + r) * MP];
                    }
                    fb.q[0] = b0.q;
                    fb.q[1] = b1.q;
                    acc.v = __builtin_amdgcn_wmma_f32_16x16x32_bf16(
                        false, fa.v, false, fb.v, (short)0, acc.v, false, false);
                }
                // epilogue: C layout -> lane col m, rows kb..kb+7 contiguous in [b][m][k]
                int m  = mt * 16 + lm;
                int kb = kt * 16 + lhi * 8;
                Q8 dq; dq.q = *(const uint4v*)(DtY + ((size_t)b * MR + m) * KP + kb);
                Q8 xq;
                #pragma unroll
                for (int r = 0; r < 8; ++r) {
                    float z  = acc.f[r] + bf2f(dq.h[r]);
                    float ab = fabsf(z) - lam;
                    xq.h[r]  = f2bf((ab > 0.f) ? copysignf(ab, z) : 0.f);
                }
                *(uint4v*)(XnB + (size_t)m * KP + kb) = xq.q;
            }
            wait_async0();       // own wave's DMAs for stage s+1 landed
            __syncthreads();     // all waves done: compute(s) + DMA(s+1)
        }

        // -------- phase B: y = (1+tt)*x_new - tt*x_old, into LDS --------
        __threadfence();
        __syncthreads();
        float omt = 1.f + tt;
        const int NCH = KP / 8;                       // 82 b128 chunks per m-row
        for (int idx = tid; idx < MR * NCH; idx += NTH) {
            int m = idx / NCH, c8 = idx - m * NCH;
            int k = c8 * 8;
            Q8 qn, qo;
            qn.q = *(const uint4v*)(XnB + (size_t)m * KP + k);
            qo.q = *(const uint4v*)(XoB + (size_t)m * KP + k);
            #pragma unroll
            for (int r = 0; r < 8; ++r) {
                float y = omt * bf2f(qn.h[r]) - tt * bf2f(qo.h[r]);
                Ylds[(size_t)(k + r) * MP + m] = f2bf(y);
            }
        }
        __syncthreads();
    }
}

__global__ void k_out(const unsigned short* __restrict__ Xf, float* __restrict__ out) {
    size_t idx = (size_t)blockIdx.x * blockDim.x + threadIdx.x;
    if (idx >= (size_t)NB * NK * MM) return;
    int m = (int)(idx % MM);
    size_t r = idx / MM;
    int k = (int)(r % NK);
    int b = (int)(r / NK);
    out[idx] = bf2f(Xf[((size_t)b * MR + m) * KP + k]);
}

// ---------------- launcher ----------------

extern "C" void kernel_launch(void* const* d_in, const int* in_sizes, int n_in,
                              void* d_out, int out_size, void* d_ws, size_t ws_size,
                              hipStream_t stream) {
    (void)in_sizes; (void)n_in; (void)out_size; (void)ws_size;
    const float* x   = (const float*)d_in[0];
    const float* Drr = (const float*)d_in[1];
    const float* Dth = (const float*)d_in[2];

    char* ws = (char*)d_ws;
    size_t off = 0;
    auto take = [&](size_t bytes) -> char* {
        char* p = ws + off;
        off = (off + bytes + 255) & ~(size_t)255;
        return p;
    };
    float*          D    = (float*)take((size_t)TT * KP * 4);
    float*          DtD  = (float*)take((size_t)NK * NK * 4);
    float*          scal = (float*)take(256);
    unsigned short* Abf  = (unsigned short*)take((size_t)KP * JP * 2);
    unsigned short* Xa   = (unsigned short*)take((size_t)NB * MR * KP * 2);
    unsigned short* Xb   = (unsigned short*)take((size_t)NB * MR * KP * 2);
    unsigned short* DtY  = (unsigned short*)d_out;   // scratch in d_out (dead before k_out)

    k_build_D<<<1, KP, 0, stream>>>(Drr, Dth, D);
    k_DtD<<<(NK * NK + 255) / 256, 256, 0, stream>>>(D, DtD);
    k_power<<<1, 1024, 0, stream>>>(DtD, scal);
    k_buildA<<<(KP * JP + 255) / 256, 256, 0, stream>>>(DtD, scal, Abf);
    k_DtY<<<NB * 10, 256, 0, stream>>>(x, D, scal, DtY);
    k_fista<<<NB, NTH, 0, stream>>>(Abf, DtY, Xa, Xb, scal);
    k_out<<<(int)(((size_t)NB * NK * MM + 255) / 256), 256, 0, stream>>>(Xb, (float*)d_out);
}
